// EncoderDecoder_46265387713233
// MI455X (gfx1250) — matrix operand order repaired and checked by reading, not compiled, not run on hardware
//
#include <hip/hip_runtime.h>
#include <hip/hip_bf16.h>

typedef __bf16 bf16_t;
typedef __attribute__((ext_vector_type(16))) __bf16 v16bf;
typedef __attribute__((ext_vector_type(8)))  __bf16 v8bf;
typedef __attribute__((ext_vector_type(8)))  float  v8f;

// ---------------------------------------------------------------------------
// WMMA helpers (gfx1250 wave32, v_wmma_f32_16x16x32_bf16)
// ---------------------------------------------------------------------------
__device__ __forceinline__ v8f wmma_bf16(v16bf a, v16bf b, v8f c) {
    // (neg_a, A, neg_b, B, c_mod, C, reuse_a, reuse_b)
    return __builtin_amdgcn_wmma_f32_16x16x32_bf16(false, a, false, b,
                                                   (short)0, c, false, false);
}

// A-matrix 16x32 bf16, row-major source with leading dim ld.
// Lane layout: lane L holds row M=L%16; contiguous K
// {base..base+15} with base = (L>=16 ? 16 : 0).
__device__ __forceinline__ v16bf load_a16x32(const bf16_t* __restrict__ tile,
                                             int ld, int lane) {
    const int row = lane & 15;
    const int cb  = (lane >> 4) * 16;
    const bf16_t* p = tile + (size_t)row * ld + cb;
    v8bf c0 = *(const v8bf*)(p);
    v8bf c1 = *(const v8bf*)(p + 8);
    return __builtin_shufflevector(c0, c1, 0,1,2,3,4,5,6,7,8,9,10,11,12,13,14,15);
}

// B-matrix 32x16 bf16 where B = W^T and W is row-major [N,K] with leading dim ld.
// Lane n holds column n of B = row n of W: 16 contiguous bf16 at
// k-offset (lane>=16 ? 16 : 0).  wtile = &W[ntile*16][k0].
__device__ __forceinline__ v16bf load_b32x16(const bf16_t* __restrict__ wtile,
                                             int ld, int lane) {
    const bf16_t* p = wtile + (size_t)(lane & 15) * ld + ((lane >> 4) * 16);
    return *(const v16bf*)p;
}

__device__ __forceinline__ float sigmoidf_(float x) {
    return 1.0f / (1.0f + __expf(-x));
}
__device__ __forceinline__ float tanhf_(float x) {
    float t = __expf(2.0f * x);
    return (t - 1.0f) / (t + 1.0f);
}

// ---------------------------------------------------------------------------
// fp32 -> bf16 conversion with row/col zero-padding
// ---------------------------------------------------------------------------
__global__ void convert_pad_kernel(const float* __restrict__ src,
                                   bf16_t* __restrict__ dst,
                                   int src_rows, int src_cols,
                                   int dst_rows, int dst_cols) {
    size_t n = (size_t)dst_rows * dst_cols;
    for (size_t i = (size_t)blockIdx.x * blockDim.x + threadIdx.x; i < n;
         i += (size_t)gridDim.x * blockDim.x) {
        int r = (int)(i / dst_cols);
        int c = (int)(i % dst_cols);
        float v = (r < src_rows && c < src_cols)
                      ? src[(size_t)r * src_cols + c] : 0.0f;
        dst[i] = (bf16_t)v;
    }
}

// ---------------------------------------------------------------------------
// Encoder GEMM:  C = act(A[M,K] @ W[N,K]^T + bias)
//   mode 0:  relu  -> bf16 C
//   mode 1:  mu -> z = eps*exp(0.5*mu)+mu -> bf16 C  (VAE reparameterization)
// Block = 256 threads (8 waves). Wave w owns a 16-row M tile x 64 cols.
// Software-pipelined K loop: loads for k+32 are issued before WMMAs for k so
// the scheduler can use partial s_wait_loadcnt instead of full serialization.
// ---------------------------------------------------------------------------
__global__ void gemm_bias_act_kernel(const bf16_t* __restrict__ A,
                                     const bf16_t* __restrict__ W,
                                     const float*  __restrict__ bias,
                                     const float*  __restrict__ eps,
                                     bf16_t* __restrict__ C,
                                     int M, int N, int K, int mode) {
    const int lane = threadIdx.x & 31;
    const int wid  = threadIdx.x >> 5;
    const int mt   = blockIdx.y * 8 + wid;     // 16-row tile index
    const int n0   = blockIdx.x * 64;
    if (mt * 16 >= M) return;

    const bf16_t* arow = A + (size_t)mt * 16 * K;
    const bf16_t* w0 = W + (size_t)(n0 +  0) * K;
    const bf16_t* w1 = W + (size_t)(n0 + 16) * K;
    const bf16_t* w2 = W + (size_t)(n0 + 32) * K;
    const bf16_t* w3 = W + (size_t)(n0 + 48) * K;

    // prologue: k = 0
    v16bf a  = load_a16x32(arow, K, lane);
    v16bf b0 = load_b32x16(w0, K, lane);
    v16bf b1 = load_b32x16(w1, K, lane);
    v16bf b2 = load_b32x16(w2, K, lane);
    v16bf b3 = load_b32x16(w3, K, lane);

    v8f acc0 = {}, acc1 = {}, acc2 = {}, acc3 = {};
    for (int k = 32; k < K; k += 32) {
        __builtin_prefetch(w0 + k + 96, 0, 1);
        __builtin_prefetch(w2 + k + 96, 0, 1);
        v16bf an  = load_a16x32(arow + k, K, lane);
        v16bf c0  = load_b32x16(w0 + k, K, lane);
        v16bf c1  = load_b32x16(w1 + k, K, lane);
        v16bf c2  = load_b32x16(w2 + k, K, lane);
        v16bf c3  = load_b32x16(w3 + k, K, lane);
        acc0 = wmma_bf16(a, b0, acc0);
        acc1 = wmma_bf16(a, b1, acc1);
        acc2 = wmma_bf16(a, b2, acc2);
        acc3 = wmma_bf16(a, b3, acc3);
        a = an; b0 = c0; b1 = c1; b2 = c2; b3 = c3;
    }
    acc0 = wmma_bf16(a, b0, acc0);
    acc1 = wmma_bf16(a, b1, acc1);
    acc2 = wmma_bf16(a, b2, acc2);
    acc3 = wmma_bf16(a, b3, acc3);

    v8f accs[4] = {acc0, acc1, acc2, acc3};
    const int moff = (lane >> 4) * 8;
    const int colb = lane & 15;
#pragma unroll
    for (int j = 0; j < 4; ++j) {
        int col = n0 + j * 16 + colb;
        float bs = bias[col];
#pragma unroll
        for (int v = 0; v < 8; ++v) {
            int m = mt * 16 + moff + v;
            float o = accs[j][v] + bs;
            if (mode == 0) {
                o = fmaxf(o, 0.0f);
                C[(size_t)m * N + col] = (bf16_t)o;
            } else {
                float e = eps[(size_t)m * N + col];
                float z = e * __expf(0.5f * o) + o;
                C[(size_t)m * N + col] = (bf16_t)z;
            }
        }
    }
}

// ---------------------------------------------------------------------------
// GRU decoder: each block owns BM=32 batch rows, keeps x_t and h resident in
// LDS across all T=128 steps.  Weights stream from L2 every step.
//   LDS: xt[32][512] + h[2][32][512] + out[32][64]  (bf16)  = 100 KB
//   grid = 4096/32 = 128 blocks -> fills the chip.
// ---------------------------------------------------------------------------
#define DEC_BM  32
#define DEC_H   512
#define DEC_T   128
#define DEC_V   42
#define DEC_VP  64      // padded fc2 K / out-buffer width

__global__ void gru_decoder_kernel(const bf16_t* __restrict__ z,     // [B,512]
                                   const bf16_t* __restrict__ wih,   // [1536,512]
                                   const bf16_t* __restrict__ whh,   // [1536,512]
                                   const float*  __restrict__ bih,   // [1536]
                                   const float*  __restrict__ bhh,   // [1536]
                                   const bf16_t* __restrict__ fc1w,  // [48,512] row-padded
                                   const float*  __restrict__ fc1b,  // [42]
                                   const bf16_t* __restrict__ fc2w,  // [512,64] col-padded
                                   const float*  __restrict__ fc2b,  // [512]
                                   float* __restrict__ out) {        // [B,T,V]
    extern __shared__ bf16_t smem[];
    bf16_t* xt   = smem;                                 // [BM][H]
    bf16_t* hbuf = smem + DEC_BM * DEC_H;                // 2 x [BM][H]
    bf16_t* obuf = smem + 3 * DEC_BM * DEC_H;            // [BM][VP]

    const int tid  = threadIdx.x;
    const int lane = tid & 31;
    const int wid  = tid >> 5;
    const int nw   = blockDim.x >> 5;                    // 8 waves
    const int row0 = blockIdx.x * DEC_BM;

    // ---- async-LDS staging of x_0 = z tile (gfx1250 GLOBAL_LOAD_ASYNC_TO_LDS)
    {
        const bf16_t* gz = z + (size_t)row0 * DEC_H;
        // 32*512 bf16 = 32 KB = 2048 x 16-byte chunks; 256 threads -> 8 iters
        for (int c = tid; c < (DEC_BM * DEC_H) / 8; c += blockDim.x) {
            unsigned lds_off = (unsigned)(size_t)(xt + c * 8);
            unsigned long long gaddr = (unsigned long long)(size_t)(gz + c * 8);
            asm volatile("global_load_async_to_lds_b128 %0, %1, off"
                         :: "v"(lds_off), "v"(gaddr) : "memory");
        }
    }
    // h_0 = 0, out buffer = 0 (keeps pad cols 48..63 zero)
    for (int i = tid; i < DEC_BM * DEC_H; i += blockDim.x) {
        hbuf[i] = (bf16_t)0.0f;
        hbuf[DEC_BM * DEC_H + i] = (bf16_t)0.0f;
    }
    for (int i = tid; i < DEC_BM * DEC_VP; i += blockDim.x)
        obuf[i] = (bf16_t)0.0f;
    asm volatile("s_wait_asynccnt 0x0" ::: "memory");
    __syncthreads();

    int cur = 0;
    for (int t = 0; t < DEC_T; ++t) {
        bf16_t* hc = hbuf + cur * DEC_BM * DEC_H;
        bf16_t* hn = hbuf + (1 - cur) * DEC_BM * DEC_H;

        // ---- phase 1: fused GRU gates -> h_new ------------------------------
        // 2 m-tiles x 32 n-tiles; per tile 6 accumulators over K=512.
        // Software-pipelined: next k's 2 A (LDS) + 6 B (L2) loads issued
        // before the 6 WMMAs of the current k.
        for (int tile = wid; tile < (DEC_BM / 16) * 32; tile += nw) {
            int mt = tile >> 5;
            int nt = tile & 31;
            const bf16_t* xrow = xt + mt * 16 * DEC_H;
            const bf16_t* hrow = hc + mt * 16 * DEC_H;
            const bf16_t* wr = wih + (size_t)(0 * DEC_H + nt * 16) * DEC_H;
            const bf16_t* wz = wih + (size_t)(1 * DEC_H + nt * 16) * DEC_H;
            const bf16_t* wn = wih + (size_t)(2 * DEC_H + nt * 16) * DEC_H;
            const bf16_t* ur = whh + (size_t)(0 * DEC_H + nt * 16) * DEC_H;
            const bf16_t* uz = whh + (size_t)(1 * DEC_H + nt * 16) * DEC_H;
            const bf16_t* un = whh + (size_t)(2 * DEC_H + nt * 16) * DEC_H;

            v16bf ax = load_a16x32(xrow, DEC_H, lane);
            v16bf ah = load_a16x32(hrow, DEC_H, lane);
            v16bf br = load_b32x16(wr, DEC_H, lane);
            v16bf bz = load_b32x16(wz, DEC_H, lane);
            v16bf bn = load_b32x16(wn, DEC_H, lane);
            v16bf crr = load_b32x16(ur, DEC_H, lane);
            v16bf czz = load_b32x16(uz, DEC_H, lane);
            v16bf cnn = load_b32x16(un, DEC_H, lane);

            v8f air = {}, aiz = {}, ain = {}, ahr = {}, ahz = {}, ahn = {};
            for (int k = 32; k < DEC_H; k += 32) {
                __builtin_prefetch(wr + k + 96, 0, 1);
                __builtin_prefetch(ur + k + 96, 0, 1);
                v16bf ax2 = load_a16x32(xrow + k, DEC_H, lane);
                v16bf ah2 = load_a16x32(hrow + k, DEC_H, lane);
                v16bf br2 = load_b32x16(wr + k, DEC_H, lane);
                v16bf bz2 = load_b32x16(wz + k, DEC_H, lane);
                v16bf bn2 = load_b32x16(wn + k, DEC_H, lane);
                v16bf cr2 = load_b32x16(ur + k, DEC_H, lane);
                v16bf cz2 = load_b32x16(uz + k, DEC_H, lane);
                v16bf cn2 = load_b32x16(un + k, DEC_H, lane);
                air = wmma_bf16(ax, br, air);
                aiz = wmma_bf16(ax, bz, aiz);
                ain = wmma_bf16(ax, bn, ain);
                ahr = wmma_bf16(ah, crr, ahr);
                ahz = wmma_bf16(ah, czz, ahz);
                ahn = wmma_bf16(ah, cnn, ahn);
                ax = ax2; ah = ah2;
                br = br2; bz = bz2; bn = bn2;
                crr = cr2; czz = cz2; cnn = cn2;
            }
            air = wmma_bf16(ax, br, air);
            aiz = wmma_bf16(ax, bz, aiz);
            ain = wmma_bf16(ax, bn, ain);
            ahr = wmma_bf16(ah, crr, ahr);
            ahz = wmma_bf16(ah, czz, ahz);
            ahn = wmma_bf16(ah, cnn, ahn);

            const int col  = nt * 16 + (lane & 15);
            const int moff = (lane >> 4) * 8;
            const float bir = bih[col],             bhr = bhh[col];
            const float biz = bih[col + DEC_H],     bhz = bhh[col + DEC_H];
            const float bin = bih[col + 2 * DEC_H], bhn = bhh[col + 2 * DEC_H];
#pragma unroll
            for (int v = 0; v < 8; ++v) {
                int m = mt * 16 + moff + v;
                float r  = sigmoidf_((air[v] + bir) + (ahr[v] + bhr));
                float zg = sigmoidf_((aiz[v] + biz) + (ahz[v] + bhz));
                float nn = tanhf_((ain[v] + bin) + r * (ahn[v] + bhn));
                float ho = (float)hc[m * DEC_H + col];
                hn[m * DEC_H + col] = (bf16_t)((1.0f - zg) * nn + zg * ho);
            }
        }
        __syncthreads();

        // ---- phase 2: out = relu(h_new @ fc1^T + b1) -> global + LDS --------
        for (int tile = wid; tile < (DEC_BM / 16) * 3; tile += nw) {
            int mt = tile / 3;
            int nt = tile % 3;
            const bf16_t* arow = hn + mt * 16 * DEC_H;
            const bf16_t* wrow = fc1w + (size_t)nt * 16 * DEC_H;
            v16bf a = load_a16x32(arow, DEC_H, lane);
            v16bf b = load_b32x16(wrow, DEC_H, lane);
            v8f acc = {};
            for (int k = 32; k < DEC_H; k += 32) {
                v16bf a2 = load_a16x32(arow + k, DEC_H, lane);
                v16bf b2 = load_b32x16(wrow + k, DEC_H, lane);
                acc = wmma_bf16(a, b, acc);
                a = a2; b = b2;
            }
            acc = wmma_bf16(a, b, acc);

            const int col  = nt * 16 + (lane & 15);
            const int moff = (lane >> 4) * 8;
            const float bs = (col < DEC_V) ? fc1b[col] : 0.0f;
#pragma unroll
            for (int v = 0; v < 8; ++v) {
                int m = mt * 16 + moff + v;
                float o = fmaxf(acc[v] + bs, 0.0f);
                obuf[m * DEC_VP + col] = (bf16_t)o;
                if (col < DEC_V)
                    out[((size_t)(row0 + m) * DEC_T + t) * DEC_V + col] = o;
            }
        }
        __syncthreads();

        // ---- phase 3: x_{t+1} = relu(out @ fc2^T + b2) ----------------------
        for (int tile = wid; tile < (DEC_BM / 16) * 32; tile += nw) {
            int mt = tile >> 5;
            int nt = tile & 31;
            const bf16_t* arow = obuf + mt * 16 * DEC_VP;
            const bf16_t* wrow = fc2w + (size_t)nt * 16 * DEC_VP;
            v16bf a0 = load_a16x32(arow, DEC_VP, lane);
            v16bf b0 = load_b32x16(wrow, DEC_VP, lane);
            v16bf a1 = load_a16x32(arow + 32, DEC_VP, lane);
            v16bf b1 = load_b32x16(wrow + 32, DEC_VP, lane);
            v8f acc = {};
            acc = wmma_bf16(a0, b0, acc);
            acc = wmma_bf16(a1, b1, acc);

            const int col  = nt * 16 + (lane & 15);
            const int moff = (lane >> 4) * 8;
            const float bs = fc2b[col];
#pragma unroll
            for (int v = 0; v < 8; ++v) {
                int m = mt * 16 + moff + v;
                xt[m * DEC_H + col] = (bf16_t)fmaxf(acc[v] + bs, 0.0f);
            }
        }
        __syncthreads();
        cur ^= 1;
    }
}

// ---------------------------------------------------------------------------
// Host launcher
// ---------------------------------------------------------------------------
extern "C" void kernel_launch(void* const* d_in, const int* in_sizes, int n_in,
                              void* d_out, int out_size, void* d_ws, size_t ws_size,
                              hipStream_t stream) {
    constexpr int B = 4096, FP = 4860, FPP = 4864, E = 512, H = 512, V = 42;
    constexpr int H1 = 2048, H2 = 1024, H3 = 512, G3 = 1536;

    const float* x    = (const float*)d_in[0];
    const float* eps  = (const float*)d_in[2];
    const float* w1   = (const float*)d_in[3];
    const float* b1   = (const float*)d_in[4];
    const float* w2   = (const float*)d_in[5];
    const float* b2   = (const float*)d_in[6];
    const float* w3   = (const float*)d_in[7];
    const float* b3   = (const float*)d_in[8];
    const float* w41  = (const float*)d_in[9];
    const float* b41  = (const float*)d_in[10];
    const float* wih  = (const float*)d_in[11];
    const float* whh  = (const float*)d_in[12];
    const float* bih  = (const float*)d_in[13];
    const float* bhh  = (const float*)d_in[14];
    const float* fc1w = (const float*)d_in[15];
    const float* fc1b = (const float*)d_in[16];
    const float* fc2w = (const float*)d_in[17];
    const float* fc2b = (const float*)d_in[18];

    char* ws = (char*)d_ws;
    size_t off = 0;
    auto alloc = [&](size_t elems) -> bf16_t* {
        bf16_t* p = (bf16_t*)(ws + off);
        off += (elems * sizeof(bf16_t) + 255) & ~(size_t)255;
        return p;
    };
    bf16_t* xb    = alloc((size_t)B * FPP);
    bf16_t* w1b   = alloc((size_t)H1 * FPP);
    bf16_t* w2b   = alloc((size_t)H2 * H1);
    bf16_t* w3b   = alloc((size_t)H3 * H2);
    bf16_t* w4b   = alloc((size_t)E * H3);
    bf16_t* wihb  = alloc((size_t)G3 * E);
    bf16_t* whhb  = alloc((size_t)G3 * H);
    bf16_t* fc1wb = alloc((size_t)48 * H);
    bf16_t* fc2wb = alloc((size_t)H * 64);
    bf16_t* h1b   = alloc((size_t)B * H1);
    bf16_t* h2b   = alloc((size_t)B * H2);
    bf16_t* h3b   = alloc((size_t)B * H3);
    bf16_t* zb    = alloc((size_t)B * E);

    auto cvt = [&](const float* src, bf16_t* dst, int sr, int sc, int dr, int dc) {
        size_t n = (size_t)dr * dc;
        size_t want = (n + 255) / 256;
        int blocks = (int)(want < 4096 ? want : 4096);
        convert_pad_kernel<<<blocks, 256, 0, stream>>>(src, dst, sr, sc, dr, dc);
    };

    // --- fp32 -> bf16 (with padding) ---
    cvt(x,    xb,    B,  FP, B,  FPP);
    cvt(w1,   w1b,   H1, FP, H1, FPP);
    cvt(w2,   w2b,   H2, H1, H2, H1);
    cvt(w3,   w3b,   H3, H2, H3, H2);
    cvt(w41,  w4b,   E,  H3, E,  H3);
    cvt(wih,  wihb,  G3, E,  G3, E);
    cvt(whh,  whhb,  G3, H,  G3, H);
    cvt(fc1w, fc1wb, V,  H,  48, H);     // pad rows 42 -> 48 with zeros
    cvt(fc2w, fc2wb, H,  V,  H,  64);    // pad cols 42 -> 64 with zeros

    // --- encoder: 4 GEMM layers ---
    dim3 blk(256, 1, 1);
    gemm_bias_act_kernel<<<dim3(H1 / 64, B / 128), blk, 0, stream>>>(
        xb, w1b, b1, nullptr, h1b, B, H1, FPP, 0);
    gemm_bias_act_kernel<<<dim3(H2 / 64, B / 128), blk, 0, stream>>>(
        h1b, w2b, b2, nullptr, h2b, B, H2, H1, 0);
    gemm_bias_act_kernel<<<dim3(H3 / 64, B / 128), blk, 0, stream>>>(
        h2b, w3b, b3, nullptr, h3b, B, H3, H2, 0);
    // layer 4 + reparameterization epilogue: z = eps*exp(0.5*mu)+mu
    gemm_bias_act_kernel<<<dim3(E / 64, B / 128), blk, 0, stream>>>(
        h3b, w4b, b41, eps, zb, B, E, H3, 1);

    // --- GRU decoder: 32 rows/block resident in LDS for all 128 steps ---
    size_t smem = (size_t)(3 * DEC_BM * DEC_H + DEC_BM * DEC_VP) * sizeof(bf16_t);
    gru_decoder_kernel<<<dim3(B / DEC_BM), blk, smem, stream>>>(
        zb, wihb, whhb, bih, bhh, fc1wb, fc1b, fc2wb, fc2b, (float*)d_out);
}
